// EdgeUpdate_88991722373554
// MI455X (gfx1250) — matrix operand order repaired and checked by reading, not compiled, not run on hardware
//
#include <hip/hip_runtime.h>

typedef __attribute__((ext_vector_type(16))) _Float16 v16h;
typedef __attribute__((ext_vector_type(8)))  float    v8f;

#define D_FEAT   64
#define K1PAD    160   // 129 padded to 5 K-tiles of 32
#define N1       128
#define N2       64
#define EPB      64    // edges per block
#define THREADS  128   // 4 wave32s
#define NFRAG1   40    // layer1: 5 kt x 8 nt
#define NFRAG2   16    // layer2: 4 kt x 4 nt
#define NFRAGS   (NFRAG1 + NFRAG2)
#define WFRAG_BYTES (NFRAGS * 1024)   // 56 KB: 32 lanes x 16 halfs per frag

static __device__ __forceinline__ v8f wmma16x16x32(v16h a, v16h b, v8f c) {
  // (neg_a, A, neg_b, B, c_mod, C, reuse_a, reuse_b)
  return __builtin_amdgcn_wmma_f32_16x16x32_f16(false, a, false, b, (short)0, c,
                                                false, false);
}

union AFrag {
  v16h v;
  struct { uint4 lo, hi; } u;
};

union H8 {
  _Float16 h[8];
  uint4 u;
};

union H16 {
  _Float16 h[16];
  uint4 u[2];
};

// ---------------------------------------------------------------------------
// Pre-pack kernels (run once per launch; cost ~microseconds)
// ---------------------------------------------------------------------------

// Pack W1/W2 (fp32) into f16 WMMA B-fragments, one fragment per block.
// Fragment layout: frag*512 halfs, lane*16 halfs per lane (contiguous K order, same as the A loads:
// lanes 0-15 hold K=0..15 of the tile, element i <-> K=kb+i).
__global__ __launch_bounds__(32) void
pack_w_frags(const float* __restrict__ W1, const float* __restrict__ W2,
             _Float16* __restrict__ wf) {
  const int id = blockIdx.x;     // 0..55
  const int l  = threadIdx.x;    // 0..31
  const int hi = l >> 4;
  const int nl = l & 15;
  H16 t;
  if (id < NFRAG1) {
    const int kt = id >> 3, nt = id & 7;
    const int n  = nt * 16 + nl;
    const int kb = kt * 32 + hi * 16;
    const float* wrow = W1 + n * 129;
#pragma unroll
    for (int i = 0; i < 16; ++i) {
      const int k    = kb + i;
      const int ksrc = (k < 128) ? (k + 1) : 0;  // k==128 -> weight col 0
      float w = wrow[ksrc];
      w = (k <= 128) ? w : 0.0f;                 // zero the K padding
      t.h[i] = (_Float16)w;
    }
  } else {
    const int f  = id - NFRAG1;
    const int kt = f >> 2, nt = f & 3;
    const int n  = nt * 16 + nl;
    const int kb = kt * 32 + hi * 16;
    const float* wrow = W2 + n * 128;
#pragma unroll
    for (int i = 0; i < 16; ++i) t.h[i] = (_Float16)wrow[kb + i];
  }
  uint4* dst = (uint4*)(wf + id * 512 + l * 16);
  dst[0] = t.u[0];
  dst[1] = t.u[1];
}

// Elementwise fp32 -> f16 of the node-feature table (8 elems/thread).
__global__ __launch_bounds__(256) void
pack_x_f16(const float* __restrict__ x, _Float16* __restrict__ xh, int n) {
  const int base = (blockIdx.x * 256 + threadIdx.x) * 8;
  if (base + 8 <= n) {
    float4 f0 = *(const float4*)(x + base);
    float4 f1 = *(const float4*)(x + base + 4);
    H8 t;
    t.h[0] = (_Float16)f0.x; t.h[1] = (_Float16)f0.y;
    t.h[2] = (_Float16)f0.z; t.h[3] = (_Float16)f0.w;
    t.h[4] = (_Float16)f1.x; t.h[5] = (_Float16)f1.y;
    t.h[6] = (_Float16)f1.z; t.h[7] = (_Float16)f1.w;
    *(uint4*)(xh + base) = t.u;
  } else {
    for (int i = base; i < n; ++i) xh[i] = (_Float16)x[i];
  }
}

// ---------------------------------------------------------------------------
// Main kernel (packed path): f16 gather copy + fragment loads + WMMA
// ---------------------------------------------------------------------------
__global__ __launch_bounds__(THREADS) void
edge_mlp_wmma_packed(const _Float16* __restrict__ xh,  // [N_NODES, 64] f16
                     const int*      __restrict__ ei,  // [2, E]
                     const float*    __restrict__ ew,  // [E, 1]
                     const _Float16* __restrict__ wf,  // packed B-fragments
                     const float*    __restrict__ b1,
                     const float*    __restrict__ b2,
                     const float*    __restrict__ W3,  // [1, 64]
                     const float*    __restrict__ b3,
                     float* __restrict__ out, int E) {
  // h1 feature order: [0..63]=src, [64..127]=tgt, [128]=w, [129..159]=0
  __shared__ __align__(16) _Float16 h1[EPB * K1PAD]; // 20 KB
  __shared__ __align__(16) _Float16 h2[EPB * N1];    // 16 KB
  __shared__ __align__(16) _Float16 h3[EPB * N2];    //  8 KB

  const int tid = threadIdx.x;

  // ---------------- Phase 0: f16 gather copy into LDS ----------------
  {
    const int e_local = tid >> 1;          // 0..63
    const int p       = tid & 1;           // 0 = src half, 1 = tgt half
    int e  = blockIdx.x * EPB + e_local;
    int ec = (e < E) ? e : (E - 1);

    const int node = ei[p * E + ec];
    const uint4* src = (const uint4*)(xh + (long)node * D_FEAT); // 128 B row
    uint4* dst = (uint4*)(h1 + e_local * K1PAD + p * D_FEAT);
#pragma unroll
    for (int c = 0; c < 8; ++c) dst[c] = src[c];     // 8 x b128

    if (p == 0) {                          // weight @128, zeros 129..143
      H8 t;
#pragma unroll
      for (int j = 0; j < 8; ++j) t.h[j] = (_Float16)0.0f;
      t.h[0] = (_Float16)ew[ec];
      *((uint4*)(h1 + e_local * K1PAD + 128)) = t.u;
      uint4 z = make_uint4(0u, 0u, 0u, 0u);
      *((uint4*)(h1 + e_local * K1PAD + 136)) = z;
    } else {                               // zeros 144..159
      uint4 z = make_uint4(0u, 0u, 0u, 0u);
      *((uint4*)(h1 + e_local * K1PAD + 144)) = z;
      *((uint4*)(h1 + e_local * K1PAD + 152)) = z;
    }
  }
  __syncthreads();

  const int lane = tid & 31;
  const int wave = tid >> 5;
  const int m0   = wave * 16;
  const int hi   = lane >> 4;
  const int nl   = lane & 15;
  const uint4* wfl = (const uint4*)(wf + lane * 16); // this lane's frag slice

  // ---------------- Layer 1: [16 x 160] x [160 x 128] ----------------
  v8f acc1[8];
#pragma unroll
  for (int nt = 0; nt < 8; ++nt) acc1[nt] = v8f{};

  {
    const _Float16* arow = h1 + (m0 + nl) * K1PAD;   // A: M = lane&15
#pragma unroll
    for (int kt = 0; kt < 5; ++kt) {
      AFrag a;
      const _Float16* ap = arow + kt * 32 + hi * 16;
      a.u.lo = *(const uint4*)(ap);        // K {0..7}|{16..23}
      a.u.hi = *(const uint4*)(ap + 8);    // K {8..15}|{24..31}
#pragma unroll
      for (int nt = 0; nt < 8; ++nt) {
        const int fid = kt * 8 + nt;
        AFrag b;
        b.u.lo = wfl[fid * 32 + 0];        // fid*32 u4 = fid*256 halfs (8 halfs per u4)
        b.u.hi = wfl[fid * 32 + 1];
        acc1[nt] = wmma16x16x32(a.v, b.v, acc1[nt]);
      }
    }
  }

  // bias + ReLU -> f16 into h2 (C layout: vgpr i -> row i+8*hi, col nl)
#pragma unroll
  for (int nt = 0; nt < 8; ++nt) {
    const int n = nt * 16 + nl;
    const float bias = b1[n];
#pragma unroll
    for (int i = 0; i < 8; ++i) {
      float v = acc1[nt][i] + bias;
      v = fmaxf(v, 0.0f);
      h2[(m0 + i + 8 * hi) * N1 + n] = (_Float16)v;
    }
  }
  // same-wave LDS ops are in-order: no barrier needed before reading h2

  // ---------------- Layer 2: [16 x 128] x [128 x 64] ----------------
  v8f acc2[4];
#pragma unroll
  for (int nt = 0; nt < 4; ++nt) acc2[nt] = v8f{};

  {
    const _Float16* arow = h2 + (m0 + nl) * N1;
#pragma unroll
    for (int kt = 0; kt < 4; ++kt) {
      AFrag a;
      const _Float16* ap = arow + kt * 32 + hi * 16;
      a.u.lo = *(const uint4*)(ap);
      a.u.hi = *(const uint4*)(ap + 8);
#pragma unroll
      for (int nt = 0; nt < 4; ++nt) {
        const int fid = NFRAG1 + kt * 4 + nt;
        AFrag b;
        b.u.lo = wfl[fid * 32 + 0];
        b.u.hi = wfl[fid * 32 + 1];
        acc2[nt] = wmma16x16x32(a.v, b.v, acc2[nt]);
      }
    }
  }

#pragma unroll
  for (int nt = 0; nt < 4; ++nt) {
    const int n = nt * 16 + nl;
    const float bias = b2[n];
#pragma unroll
    for (int i = 0; i < 8; ++i) {
      float v = acc2[nt][i] + bias;
      v = fmaxf(v, 0.0f);
      h3[(m0 + i + 8 * hi) * N2 + n] = (_Float16)v;
    }
  }

  // ---------------- Layer 3: [16 x 64] . [64] + b3 (VALU) ----------------
  if (lane < 16) {
    const int m = m0 + lane;
    const int e = blockIdx.x * EPB + m;
    const uint4* hr = (const uint4*)(h3 + m * N2);
    float s = b3[0];
#pragma unroll
    for (int c = 0; c < 8; ++c) {          // 8 x ds_load_b128
      H8 t; t.u = hr[c];
#pragma unroll
      for (int j = 0; j < 8; ++j)
        s = fmaf((float)t.h[j], W3[c * 8 + j], s);
    }
    if (e < E) out[e] = s;
  }
}

// ---------------------------------------------------------------------------
// Fallback (round-1 kernel): used only if ws_size is too small.
// ---------------------------------------------------------------------------
__global__ __launch_bounds__(THREADS) void
edge_mlp_wmma_fallback(const float* __restrict__ x,
                       const int*   __restrict__ ei,
                       const float* __restrict__ ew,
                       const float* __restrict__ W1,
                       const float* __restrict__ b1,
                       const float* __restrict__ W2,
                       const float* __restrict__ b2,
                       const float* __restrict__ W3,
                       const float* __restrict__ b3,
                       float* __restrict__ out, int E) {
  __shared__ __align__(16) _Float16 h1[EPB * K1PAD];
  __shared__ __align__(16) _Float16 h2[EPB * N1];
  __shared__ __align__(16) _Float16 h3[EPB * N2];

  const int tid = threadIdx.x;
  {
    const int e_local = tid >> 1;
    const int p       = tid & 1;
    int e  = blockIdx.x * EPB + e_local;
    int ec = (e < E) ? e : (E - 1);
    const int node = ei[p * E + ec];
    const float* xr = x + (long)node * D_FEAT;
    _Float16* dst = h1 + e_local * K1PAD + p * D_FEAT;
#pragma unroll
    for (int c = 0; c < 8; ++c) {
      float4 f0 = *(const float4*)(xr + c * 8);
      float4 f1 = *(const float4*)(xr + c * 8 + 4);
      H8 t;
      t.h[0] = (_Float16)f0.x; t.h[1] = (_Float16)f0.y;
      t.h[2] = (_Float16)f0.z; t.h[3] = (_Float16)f0.w;
      t.h[4] = (_Float16)f1.x; t.h[5] = (_Float16)f1.y;
      t.h[6] = (_Float16)f1.z; t.h[7] = (_Float16)f1.w;
      ((uint4*)dst)[c] = t.u;
    }
    if (p == 0) {
      H8 t;
#pragma unroll
      for (int j = 0; j < 8; ++j) t.h[j] = (_Float16)0.0f;
      t.h[0] = (_Float16)ew[ec];
      *((uint4*)(h1 + e_local * K1PAD + 128)) = t.u;
      uint4 z = make_uint4(0u, 0u, 0u, 0u);
      *((uint4*)(h1 + e_local * K1PAD + 136)) = z;
    } else {
      uint4 z = make_uint4(0u, 0u, 0u, 0u);
      *((uint4*)(h1 + e_local * K1PAD + 144)) = z;
      *((uint4*)(h1 + e_local * K1PAD + 152)) = z;
    }
  }
  __syncthreads();

  const int lane = tid & 31;
  const int wave = tid >> 5;
  const int m0   = wave * 16;
  const int hi   = lane >> 4;
  const int nl   = lane & 15;

  v8f acc1[8];
#pragma unroll
  for (int nt = 0; nt < 8; ++nt) acc1[nt] = v8f{};
  {
    const _Float16* arow = h1 + (m0 + nl) * K1PAD;
#pragma unroll
    for (int kt = 0; kt < 5; ++kt) {
      AFrag a;
      const _Float16* ap = arow + kt * 32 + hi * 16;
      a.u.lo = *(const uint4*)(ap);
      a.u.hi = *(const uint4*)(ap + 8);
      const int kb = kt * 32 + hi * 16;
#pragma unroll
      for (int nt = 0; nt < 8; ++nt) {
        const int n = nt * 16 + nl;
        const float* wrow = W1 + n * 129;
        v16h b;
#pragma unroll
        for (int i = 0; i < 16; ++i) {
          const int k    = kb + i;
          const int ksrc = (k < 128) ? (k + 1) : 0;
          float w = wrow[ksrc];
          w = (k <= 128) ? w : 0.0f;
          b[i] = (_Float16)w;
        }
        acc1[nt] = wmma16x16x32(a.v, b, acc1[nt]);
      }
    }
  }
#pragma unroll
  for (int nt = 0; nt < 8; ++nt) {
    const int n = nt * 16 + nl;
    const float bias = b1[n];
#pragma unroll
    for (int i = 0; i < 8; ++i) {
      float v = acc1[nt][i] + bias;
      v = fmaxf(v, 0.0f);
      h2[(m0 + i + 8 * hi) * N1 + n] = (_Float16)v;
    }
  }

  v8f acc2[4];
#pragma unroll
  for (int nt = 0; nt < 4; ++nt) acc2[nt] = v8f{};
  {
    const _Float16* arow = h2 + (m0 + nl) * N1;
#pragma unroll
    for (int kt = 0; kt < 4; ++kt) {
      AFrag a;
      const _Float16* ap = arow + kt * 32 + hi * 16;
      a.u.lo = *(const uint4*)(ap);
      a.u.hi = *(const uint4*)(ap + 8);
      const int kb = kt * 32 + hi * 16;
#pragma unroll
      for (int nt = 0; nt < 4; ++nt) {
        const int n = nt * 16 + nl;
        const float* wrow = W2 + n * 128;
        v16h b;
#pragma unroll
        for (int c = 0; c < 4; ++c) {
          float4 w4 = *(const float4*)(wrow + kb + c * 4);
          b[c * 4 + 0] = (_Float16)w4.x;
          b[c * 4 + 1] = (_Float16)w4.y;
          b[c * 4 + 2] = (_Float16)w4.z;
          b[c * 4 + 3] = (_Float16)w4.w;
        }
        acc2[nt] = wmma16x16x32(a.v, b, acc2[nt]);
      }
    }
  }
#pragma unroll
  for (int nt = 0; nt < 4; ++nt) {
    const int n = nt * 16 + nl;
    const float bias = b2[n];
#pragma unroll
    for (int i = 0; i < 8; ++i) {
      float v = acc2[nt][i] + bias;
      v = fmaxf(v, 0.0f);
      h3[(m0 + i + 8 * hi) * N2 + n] = (_Float16)v;
    }
  }

  if (lane < 16) {
    const int m = m0 + lane;
    const int e = blockIdx.x * EPB + m;
    const _Float16* hr = h3 + m * N2;
    float s = b3[0];
#pragma unroll
    for (int k = 0; k < 64; ++k) s = fmaf((float)hr[k], W3[k], s);
    if (e < E) out[e] = s;
  }
}

// ---------------------------------------------------------------------------
extern "C" void kernel_launch(void* const* d_in, const int* in_sizes, int n_in,
                              void* d_out, int out_size, void* d_ws, size_t ws_size,
                              hipStream_t stream) {
  (void)n_in; (void)out_size;
  const float* x  = (const float*)d_in[0];
  const int*   ei = (const int*)d_in[1];
  const float* ew = (const float*)d_in[2];
  const float* W1 = (const float*)d_in[3];
  const float* b1 = (const float*)d_in[4];
  const float* W2 = (const float*)d_in[5];
  const float* b2 = (const float*)d_in[6];
  const float* W3 = (const float*)d_in[7];
  const float* b3 = (const float*)d_in[8];
  float* out = (float*)d_out;

  const int E   = in_sizes[2];             // edge_weight element count == E
  const int n_x = in_sizes[0];             // node-feature element count
  const int blocks = (E + EPB - 1) / EPB;  // 12500 for E=800000

  const size_t need = (size_t)WFRAG_BYTES + (size_t)n_x * sizeof(_Float16);
  if (d_ws != nullptr && ws_size >= need) {
    _Float16* wf = (_Float16*)d_ws;
    _Float16* xh = (_Float16*)((char*)d_ws + WFRAG_BYTES);
    pack_w_frags<<<NFRAGS, 32, 0, stream>>>(W1, W2, wf);
    const int xblocks = (n_x / 8 + 255) / 256;
    pack_x_f16<<<xblocks, 256, 0, stream>>>(x, xh, n_x);
    edge_mlp_wmma_packed<<<blocks, THREADS, 0, stream>>>(xh, ei, ew, wf, b1, b2,
                                                         W3, b3, out, E);
  } else {
    edge_mlp_wmma_fallback<<<blocks, THREADS, 0, stream>>>(x, ei, ew, W1, b1,
                                                           W2, b2, W3, b3, out, E);
  }
}
